// MultiHeadAttention_28295244546620
// MI455X (gfx1250) — matrix operand order repaired and checked by reading, not compiled, not run on hardware
//
#include <hip/hip_runtime.h>
#include <hip/hip_bf16.h>

// ---------------------------------------------------------------------------
// MHA for MI455X (gfx1250), bf16 WMMA everywhere, f32 accumulate.
//   stage 0: convert query/value/W* to bf16 (scale folded into Wq)
//   stage 1: Q/K [BH,S,d] bf16 and Vt [BH,d,S] bf16 via WMMA GEMM
//   stage 2: flash attention; K/V tiles staged block-wide into LDS via
//            GLOBAL_LOAD_ASYNC_TO_LDS_B128 (ASYNCcnt).
// ---------------------------------------------------------------------------

typedef __attribute__((ext_vector_type(16))) __bf16    v16bf;
typedef __attribute__((ext_vector_type(2)))  __bf16    v2bf;
typedef __attribute__((ext_vector_type(8)))  float     v8f;
typedef __attribute__((ext_vector_type(8)))  unsigned  v8u;
typedef __attribute__((vector_size(16)))     int       i32x4;   // matches builtin param

#define S_LEN  2048
#define DMODEL 1024
#define HEADS  16
#define HDIM   64
#define BATCH  4

#if __has_builtin(__builtin_amdgcn_global_load_async_to_lds_b128)
#define HAVE_ASYNC_LDS 1
#else
#define HAVE_ASYNC_LDS 0
#endif

// ---- f32 -> bf16 helpers ---------------------------------------------------
__device__ __forceinline__ unsigned bf16b(float f) {
  unsigned u = __builtin_bit_cast(unsigned, f);
  return (u + 0x7FFFu + ((u >> 16) & 1u)) >> 16;   // RNE
}
__device__ __forceinline__ unsigned cvt2(float a, float b) {
#if __has_builtin(__builtin_amdgcn_cvt_pk_bf16_f32)
  v2bf r = __builtin_amdgcn_cvt_pk_bf16_f32(a, b);
  return __builtin_bit_cast(unsigned, r);
#else
  return bf16b(a) | (bf16b(b) << 16);
#endif
}

// ---- bf16 fragment loads (pure b128 loads, no ALU) -------------------------
// A-operand (16x32): lane half h: elems 0..7 -> K=kb+h*8+i ; 8..15 -> K=kb+16+h*8+i
__device__ __forceinline__ v16bf frag_a(const unsigned short* row, int kb, int half) {
  uint4 a = *(const uint4*)(row + kb + half * 8);
  uint4 b = *(const uint4*)(row + kb + 16 + half * 8);
  v8u u; u[0]=a.x; u[1]=a.y; u[2]=a.z; u[3]=a.w; u[4]=b.x; u[5]=b.y; u[6]=b.z; u[7]=b.w;
  return __builtin_bit_cast(v16bf, u);
}
// B-operand fed as rows of B^T: lane half h: elems 0..7 -> K=kb+h*8+i ; 8..15 -> K=kb+16+h*8+i
__device__ __forceinline__ v16bf frag_b(const unsigned short* row, int kb, int half) {
  uint4 a = *(const uint4*)(row + kb + half * 8);
  uint4 b = *(const uint4*)(row + kb + 16 + half * 8);
  v8u u; u[0]=a.x; u[1]=a.y; u[2]=a.z; u[3]=a.w; u[4]=b.x; u[5]=b.y; u[6]=b.z; u[7]=b.w;
  return __builtin_bit_cast(v16bf, u);
}

__device__ __forceinline__ v8f wmma_bf16(v16bf a, v16bf b, v8f c) {
  return __builtin_amdgcn_wmma_f32_16x16x32_bf16(false, a, false, b, (short)0, c, false, false);
}

// ---- 16B global -> LDS staging ---------------------------------------------
__device__ __forceinline__ void stage16(const unsigned short* g, unsigned short* l) {
#if HAVE_ASYNC_LDS
  __builtin_amdgcn_global_load_async_to_lds_b128(
      (__attribute__((address_space(1))) i32x4*)(g),
      (__attribute__((address_space(3))) i32x4*)(l), 0, 0);
#else
  *(uint4*)l = *(const uint4*)g;
#endif
}
__device__ __forceinline__ void wait_stage() {
#if HAVE_ASYNC_LDS
#if __has_builtin(__builtin_amdgcn_s_wait_asynccnt)
  __builtin_amdgcn_s_wait_asynccnt(0);
#else
  asm volatile("s_wait_asynccnt 0x0" ::: "memory");
#endif
#endif
}

// ---------------------------------------------------------------------------
// Stage 0: elementwise f32 -> bf16 (optionally scaled), 8 elems/thread
// ---------------------------------------------------------------------------
__global__ void __launch_bounds__(256)
cvt_kernel(const float* __restrict__ in, unsigned short* __restrict__ out,
           float scale, int n) {
  int i = (blockIdx.x * 256 + threadIdx.x) * 8;
  if (i >= n) return;
  const float4* p = (const float4*)(in + i);
  float4 a = p[0], b = p[1];
  uint4 r;
  r.x = cvt2(a.x * scale, a.y * scale);
  r.y = cvt2(a.z * scale, a.w * scale);
  r.z = cvt2(b.x * scale, b.y * scale);
  r.w = cvt2(b.z * scale, b.w * scale);
  *(uint4*)(out + i) = r;
}

// ---------------------------------------------------------------------------
// Stage 1: out(m,n) = sum_k X[m,k]*W[n,k]  (x @ W^T), bf16 in, bf16 out.
// Wave: 32(M) x 64(N); block: 8 waves = 256 rows.
// vt_mode=0: out[bh,s,k]   vt_mode=1: out[bh,k,s]
// ---------------------------------------------------------------------------
__global__ void __launch_bounds__(256)
proj_kernel(const unsigned short* __restrict__ X, const unsigned short* __restrict__ W,
            unsigned short* __restrict__ out, int vt_mode) {
  const int tid  = threadIdx.x;
  const int wave = tid >> 5;
  const int lane = tid & 31;
  const int half = lane >> 4;
  const int ln   = lane & 15;

  const int mBase = blockIdx.x * 256 + wave * 32;
  const int nBase = blockIdx.y * 64;

  v8f acc[2][4];
#pragma unroll
  for (int mt = 0; mt < 2; ++mt)
#pragma unroll
    for (int nt = 0; nt < 4; ++nt) acc[mt][nt] = v8f{};

  const unsigned short* arow0 = X + (size_t)(mBase + ln) * DMODEL;
  const unsigned short* arow1 = arow0 + (size_t)16 * DMODEL;

  for (int kb = 0; kb < DMODEL; kb += 32) {
    v16bf a0 = frag_a(arow0, kb, half);
    v16bf a1 = frag_a(arow1, kb, half);
#pragma unroll
    for (int nt = 0; nt < 4; ++nt) {
      const unsigned short* brow = W + (size_t)(nBase + nt * 16 + ln) * DMODEL;
      v16bf bfr = frag_b(brow, kb, half);
      acc[0][nt] = wmma_bf16(a0, bfr, acc[0][nt]);
      acc[1][nt] = wmma_bf16(a1, bfr, acc[1][nt]);
    }
  }

  // packed bf16 epilogue: convert row pairs with one v_cvt_pk, store lo/hi
#pragma unroll
  for (int mt = 0; mt < 2; ++mt)
#pragma unroll
    for (int nt = 0; nt < 4; ++nt) {
      int n = nBase + nt * 16 + ln;
      int hh = n >> 6, k = n & (HDIM - 1);
#pragma unroll
      for (int r = 0; r < 8; r += 2) {
        int m0 = mBase + mt * 16 + r + half * 8;
        int b = m0 >> 11, s0 = m0 & (S_LEN - 1);
        unsigned pp = cvt2(acc[mt][nt][r], acc[mt][nt][r + 1]);
        size_t i0, i1;
        if (!vt_mode) {
          size_t basei = (((size_t)(b * HEADS + hh)) * S_LEN + s0) * HDIM + k;
          i0 = basei; i1 = basei + HDIM;                 // next s
        } else {
          size_t basei = (((size_t)(b * HEADS + hh)) * HDIM + k) * S_LEN + s0;
          i0 = basei; i1 = basei + 1;                    // next s (contig in Vt)
        }
        out[i0] = (unsigned short)(pp & 0xFFFFu);
        out[i1] = (unsigned short)(pp >> 16);
      }
    }
}

// ---------------------------------------------------------------------------
// Stage 2: flash attention; one wave per 16-row q tile; kv step = 32.
// K/V tiles staged into LDS once per block (4 waves share the same head).
// Q is pre-scaled by 1/sqrt(D)*log2(e) -> softmax is pure exp2.
// ---------------------------------------------------------------------------
__global__ void __launch_bounds__(128)
attn_kernel(const unsigned short* __restrict__ Q, const unsigned short* __restrict__ K,
            const unsigned short* __restrict__ Vt, float* __restrict__ Out) {
  __shared__ alignas(16) unsigned short ktile[32 * HDIM];     // [kvrow][d]   4KB
  __shared__ alignas(16) unsigned short vtile[HDIM * 32];     // [d][kv]      4KB
  __shared__ alignas(16) unsigned short pstage[4 * 16 * 32];  // per-wave P   4KB

  const int tid  = threadIdx.x;
  const int wave = tid >> 5;
  const int lane = tid & 31;
  const int half = lane >> 4;
  const int ln   = lane & 15;

  const int bh    = blockIdx.y;                  // b*H + h
  const int qBase = (blockIdx.x * 4 + wave) * 16;

  const unsigned short* Qh = Q  + (size_t)bh * S_LEN * HDIM;
  const unsigned short* Kh = K  + (size_t)bh * S_LEN * HDIM;
  const unsigned short* Vh = Vt + (size_t)bh * HDIM * S_LEN;
  unsigned short* myp = pstage + wave * (16 * 32);

  const unsigned short* qrow = Qh + (size_t)(qBase + ln) * HDIM;
  v16bf qf0 = frag_a(qrow, 0,  half);
  v16bf qf1 = frag_a(qrow, 32, half);

  v8f o[4];
#pragma unroll
  for (int nt = 0; nt < 4; ++nt) o[nt] = v8f{};
  float m[8], l[8];
#pragma unroll
  for (int r = 0; r < 8; ++r) { m[r] = -3.0e38f; l[r] = 0.0f; }

  for (int j = 0; j < S_LEN; j += 32) {
    // ---- cooperative stage: K tile 32x64, V tile 64x32 (16B chunks) ----
#pragma unroll
    for (int c = tid; c < 256; c += 128) {       // K: chunk c = (row, 8-col grp)
      int row = c >> 3, c8 = c & 7;
      stage16(Kh + (size_t)(j + row) * HDIM + c8 * 8, ktile + c * 8);
    }
#pragma unroll
    for (int c = tid; c < 256; c += 128) {       // V: chunk c = (d-row, 8-col grp)
      int row = c >> 2, c4 = c & 3;
      stage16(Vh + (size_t)row * S_LEN + j + c4 * 8, vtile + c * 8);
    }
    wait_stage();
    __syncthreads();

    // ---- scores from LDS ----
    v8f s0 = v8f{}, s1 = v8f{};
    {
      const unsigned short* kr0 = ktile + (size_t)ln * HDIM;
      const unsigned short* kr1 = ktile + (size_t)(16 + ln) * HDIM;
      v16bf k00 = frag_b(kr0, 0,  half);
      v16bf k01 = frag_b(kr0, 32, half);
      v16bf k10 = frag_b(kr1, 0,  half);
      v16bf k11 = frag_b(kr1, 32, half);
      s0 = wmma_bf16(qf0, k00, s0);  s0 = wmma_bf16(qf1, k01, s0);
      s1 = wmma_bf16(qf0, k10, s1);  s1 = wmma_bf16(qf1, k11, s1);
    }

    // ---- online softmax ----
#pragma unroll
    for (int r = 0; r < 8; ++r) {
      float rm = fmaxf(s0[r], s1[r]);
      rm = fmaxf(rm, __shfl_xor(rm, 1, 32));
      rm = fmaxf(rm, __shfl_xor(rm, 2, 32));
      rm = fmaxf(rm, __shfl_xor(rm, 4, 32));
      rm = fmaxf(rm, __shfl_xor(rm, 8, 32));
      float mo = m[r];
      float mn = fmaxf(mo, rm);
      float alpha = exp2f(mo - mn);
      float p0 = exp2f(s0[r] - mn);
      float p1 = exp2f(s1[r] - mn);
      float rs = p0 + p1;
      rs += __shfl_xor(rs, 1, 32);
      rs += __shfl_xor(rs, 2, 32);
      rs += __shfl_xor(rs, 4, 32);
      rs += __shfl_xor(rs, 8, 32);
      l[r] = l[r] * alpha + rs;
      m[r] = mn;
#pragma unroll
      for (int nt = 0; nt < 4; ++nt) o[nt][r] *= alpha;
      int row = r + half * 8;
      unsigned pp = cvt2(p0, p1);
      myp[row * 32 + ln]      = (unsigned short)(pp & 0xFFFFu);
      myp[row * 32 + 16 + ln] = (unsigned short)(pp >> 16);
    }

    asm volatile("s_wait_dscnt 0x0" ::: "memory");

    // ---- reload P as 16x32 A fragment ----
    v16bf pf;
    {
      uint4 x0 = *(const uint4*)(myp + ln * 32 + half * 8);
      uint4 x1 = *(const uint4*)(myp + ln * 32 + 16 + half * 8);
      v8u u;
      u[0] = x0.x; u[1] = x0.y; u[2] = x0.z; u[3] = x0.w;
      u[4] = x1.x; u[5] = x1.y; u[6] = x1.z; u[7] = x1.w;
      pf = __builtin_bit_cast(v16bf, u);
    }

    // ---- O += P x V from LDS ----
#pragma unroll
    for (int nt = 0; nt < 4; ++nt) {
      const unsigned short* vrow = vtile + (size_t)(nt * 16 + ln) * 32;
      v16bf vf = frag_b(vrow, 0, half);
      o[nt] = wmma_bf16(pf, vf, o[nt]);
    }

    __syncthreads();   // protect tiles before next iteration overwrites
  }

  const int b = bh >> 4, h = bh & (HEADS - 1);
#pragma unroll
  for (int r = 0; r < 8; ++r) {
    float inv = 1.0f / l[r];
    int s = qBase + r + half * 8;
#pragma unroll
    for (int nt = 0; nt < 4; ++nt) {
      int col = h * HDIM + nt * 16 + ln;
      Out[((size_t)b * S_LEN + s) * DMODEL + col] = o[nt][r] * inv;
    }
  }
}

// ---------------------------------------------------------------------------
extern "C" void kernel_launch(void* const* d_in, const int* in_sizes, int n_in,
                              void* d_out, int out_size, void* d_ws, size_t ws_size,
                              hipStream_t stream) {
  (void)in_sizes; (void)n_in; (void)out_size; (void)ws_size;
  const float* query = (const float*)d_in[0];
  const float* value = (const float*)d_in[1];
  const float* Wq    = (const float*)d_in[2];
  const float* Wk    = (const float*)d_in[3];
  const float* Wv    = (const float*)d_in[4];
  float* out = (float*)d_out;

  const int elems = BATCH * S_LEN * DMODEL;   // 8388608
  const int wel   = DMODEL * DMODEL;          // 1048576

  unsigned short* base = (unsigned short*)d_ws;
  unsigned short* Qb  = base;                 // bf16 Q   [BH,S,d]
  unsigned short* Kb  = Qb  + elems;          // bf16 K   [BH,S,d]
  unsigned short* Vtb = Kb  + elems;          // bf16 V^T [BH,d,S]
  unsigned short* Xq  = Vtb + elems;          // bf16 query
  unsigned short* Xv  = Xq  + elems;          // bf16 value
  unsigned short* Wqb = Xv  + elems;          // bf16 Wq (pre-scaled)
  unsigned short* Wkb = Wqb + wel;
  unsigned short* Wvb = Wkb + wel;

  const float QSCALE = 0.03125f * 1.44269504088896f;  // 1/sqrt(1024)*log2(e)

  cvt_kernel<<<elems / (256 * 8), 256, 0, stream>>>(query, Xq, 1.0f, elems);
  cvt_kernel<<<elems / (256 * 8), 256, 0, stream>>>(value, Xv, 1.0f, elems);
  cvt_kernel<<<wel / (256 * 8), 256, 0, stream>>>(Wq, Wqb, QSCALE, wel);
  cvt_kernel<<<wel / (256 * 8), 256, 0, stream>>>(Wk, Wkb, 1.0f, wel);
  cvt_kernel<<<wel / (256 * 8), 256, 0, stream>>>(Wv, Wvb, 1.0f, wel);

  dim3 pgrid((BATCH * S_LEN) / 256, DMODEL / 64, 1);  // (32, 16)
  proj_kernel<<<pgrid, 256, 0, stream>>>(Xq, Wqb, Qb, 0);
  proj_kernel<<<pgrid, 256, 0, stream>>>(Xv, Wkb, Kb, 0);
  proj_kernel<<<pgrid, 256, 0, stream>>>(Xv, Wvb, Vtb, 1);

  dim3 agrid(S_LEN / 64, BATCH * HEADS, 1);           // (32, 64)
  attn_kernel<<<agrid, 128, 0, stream>>>(Qb, Kb, Vtb, out);
}
